// RNNTextClassifier_54013508714622
// MI455X (gfx1250) — matrix operand order repaired and checked by reading, not compiled, not run on hardware
//
#include <hip/hip_runtime.h>

#define DEV __device__ __forceinline__

typedef __attribute__((ext_vector_type(16))) __bf16        v16bf;
typedef __attribute__((ext_vector_type(8)))  float         v8f;
typedef __attribute__((ext_vector_type(4)))  float         v4f;
typedef __attribute__((ext_vector_type(16))) unsigned short v16u;
typedef __attribute__((ext_vector_type(8)))  unsigned short v8u;

static constexpr int BATCH = 64;
static constexpr int SEQ   = 512;
static constexpr int EMB   = 512;
static constexpr int HID   = 512;
static constexpr int NCLS  = 8;
static constexpr int MROWS = BATCH * SEQ;   // 32768

// ---- helpers -------------------------------------------------------------

DEV unsigned short f2bf(float f) {          // round-to-nearest-even f32 -> bf16
  unsigned int u = __float_as_uint(f);
  u += 0x7FFFu + ((u >> 16) & 1u);
  return (unsigned short)(u >> 16);
}
DEV v8f wmma_bf16(v16bf a, v16bf b, v8f c) {
  return __builtin_amdgcn_wmma_f32_16x16x32_bf16(false, a, false, b,
                                                 (short)0, c, false, false);
}
DEV v16bf mk_afrag(const unsigned short* row, int k0, int hi) {
  // A fragment (16-bit A 16x32): lane<16 holds K = k0+{0..15},
  // lane>=16 holds K = k0+{16..31}; hi = lane>>4.
  v16u av = *(const v16u*)(row + k0 + hi * 16);
  return __builtin_bit_cast(v16bf, av);
}
DEV v16bf mk_bfrag(const unsigned short* Wrow, int k0, int hi) {
  // B fragment (32x16): lane n = lane&15; lane<16 -> K=k0..k0+15, else +16.
  v16u bv = *(const v16u*)(Wrow + k0 + hi * 16);
  return __builtin_bit_cast(v16bf, bv);
}

// ---- embedding gather + f32->bf16 ---------------------------------------
__global__ void embed_gather_kernel(const int* __restrict__ x,
                                    const float* __restrict__ emb,
                                    unsigned short* __restrict__ dst) {
  int row = blockIdx.x;                      // 0..MROWS-1  (b*SEQ + s)
  int tok = x[row];
  const float* src = emb + (size_t)tok * EMB;
  unsigned short* d = dst + (size_t)row * EMB;
  int c = threadIdx.x * 8;                   // 64 threads * 8 = 512
  v4f f0 = *(const v4f*)(src + c);
  v4f f1 = *(const v4f*)(src + c + 4);
  v8u o;
  #pragma unroll
  for (int i = 0; i < 4; i++) { o[i] = f2bf(f0[i]); o[i + 4] = f2bf(f1[i]); }
  *(v8u*)(d + c) = o;
}

// ---- f32 -> bf16 weight conversion --------------------------------------
__global__ void cvt_bf16_kernel(const float* __restrict__ src,
                                unsigned short* __restrict__ dst, int n) {
  int i = (blockIdx.x * blockDim.x + threadIdx.x) * 8;
  if (i + 8 <= n) {
    v4f f0 = *(const v4f*)(src + i);
    v4f f1 = *(const v4f*)(src + i + 4);
    v8u o;
    #pragma unroll
    for (int j = 0; j < 4; j++) { o[j] = f2bf(f0[j]); o[j + 4] = f2bf(f1[j]); }
    *(v8u*)(dst + i) = o;
  }
}

// ---- input projection GEMM ----------------------------------------------
// out[s*BATCH + b][h] = sum_k A[b*SEQ+s][k] * W[h][k] + bias_a[h] + bias_b[h]
// A: bf16 [MROWS x 512] row-major; W: bf16 [512 x 512] row-major (N x K).
// Block = 256 threads = 8 waves; block owns a 32-row M tile, each wave a
// 64-wide N strip over both M tiles (2x4 WMMA tiles, 8 accumulators).
__global__ __launch_bounds__(256)
void gemm_inproj_kernel(const unsigned short* __restrict__ A,
                        const unsigned short* __restrict__ W,
                        const float* __restrict__ bias_a,
                        const float* __restrict__ bias_b,
                        float* __restrict__ out) {
  const int m0   = blockIdx.x * 32;
  const int lane = threadIdx.x & 31;
  const int wave = threadIdx.x >> 5;
  const int n0   = wave * 64;
  const int lm   = lane & 15;
  const int hi   = lane >> 4;

  v8f acc[2][4];
  #pragma unroll
  for (int mt = 0; mt < 2; mt++)
    #pragma unroll
    for (int j = 0; j < 4; j++) acc[mt][j] = 0.0f;

  const unsigned short* arow0 = A + (size_t)(m0 + lm) * HID;
  const unsigned short* arow1 = A + (size_t)(m0 + 16 + lm) * HID;
  for (int k0 = 0; k0 < HID; k0 += 32) {
    v16bf af0 = mk_afrag(arow0, k0, hi);
    v16bf af1 = mk_afrag(arow1, k0, hi);
    #pragma unroll
    for (int j = 0; j < 4; j++) {
      const unsigned short* wrow = W + (size_t)(n0 + j * 16 + lm) * HID;
      v16bf bf = mk_bfrag(wrow, k0, hi);
      acc[0][j] = wmma_bf16(af0, bf, acc[0][j]);
      acc[1][j] = wmma_bf16(af1, bf, acc[1][j]);
    }
  }
  #pragma unroll
  for (int j = 0; j < 4; j++) {
    int n = n0 + j * 16 + lm;
    float bsum = bias_a[n] + bias_b[n];
    #pragma unroll
    for (int mt = 0; mt < 2; mt++) {
      #pragma unroll
      for (int r = 0; r < 8; r++) {
        int m = m0 + mt * 16 + r + hi * 8;  // C layout: VGPR r -> M = m0+r (+8 hi)
        int b = m >> 9;                     // / SEQ
        int s = m & (SEQ - 1);
        out[((size_t)s * BATCH + b) * HID + n] = acc[mt][j][r] + bsum;
      }
    }
  }
}

// ---- recurrent scan (single persistent workgroup) -----------------------
// pre: f32 [SEQ][BATCH][HID].  h_{t} = tanh(pre_t + h_{t-1} @ Whh^T)
// 1024 threads = 32 waves; wave w owns N-tile w (16 cols) x 4 M-tiles.
// Split-K: two accumulator sets (k<256 / k>=256) halve the WMMA RAW chain.
template <bool STORE_SEQ, bool STORE_FINAL>
__global__ __launch_bounds__(1024)
void rnn_scan_kernel(const float* __restrict__ pre,
                     const unsigned short* __restrict__ Whh,  // bf16 [512x512]
                     unsigned short* __restrict__ out_seq,    // bf16 [MROWS x 512]
                     float* __restrict__ h_final) {           // f32 [64x512]
  __shared__ unsigned short hsh[BATCH * HID];                 // 64 KB bf16
  for (int i = threadIdx.x; i < BATCH * HID; i += 1024) hsh[i] = 0;
  __syncthreads();

  const int lane = threadIdx.x & 31;
  const int wave = threadIdx.x >> 5;   // n-tile
  const int lm   = lane & 15;
  const int hi   = lane >> 4;
  const int n    = wave * 16 + lm;
  const unsigned short* wrow = Whh + (size_t)n * HID;

  for (int t = 0; t < SEQ; t++) {
    // prefetch next step's pre slab while this step's matmul runs
    const float* ptn = pre + (size_t)(t + 1 < SEQ ? t + 1 : t) * BATCH * HID;
    __builtin_prefetch(ptn + (size_t)(hi * 8) * HID + n, 0, 3);
    __builtin_prefetch(ptn + (size_t)(32 + hi * 8) * HID + n, 0, 3);

    v8f acc0[4], acc1[4];
    #pragma unroll
    for (int j = 0; j < 4; j++) { acc0[j] = 0.0f; acc1[j] = 0.0f; }

    for (int k0 = 0; k0 < HID / 2; k0 += 32) {
      v16bf bfa = mk_bfrag(wrow, k0, hi);
      v16bf bfb = mk_bfrag(wrow, k0 + 256, hi);
      #pragma unroll
      for (int mt = 0; mt < 4; mt++) {
        const unsigned short* hr = hsh + (size_t)(mt * 16 + lm) * HID;
        acc0[mt] = wmma_bf16(mk_afrag(hr, k0, hi), bfa, acc0[mt]);
        acc1[mt] = wmma_bf16(mk_afrag(hr, k0 + 256, hi), bfb, acc1[mt]);
      }
    }

    const float* pt = pre + (size_t)t * BATCH * HID;
    unsigned short hnew[4][8];
    #pragma unroll
    for (int mt = 0; mt < 4; mt++) {
      #pragma unroll
      for (int r = 0; r < 8; r++) {
        int b = mt * 16 + r + hi * 8;
        float v = tanhf(pt[(size_t)b * HID + n] + acc0[mt][r] + acc1[mt][r]);
        hnew[mt][r] = f2bf(v);
        if (STORE_FINAL && t == SEQ - 1) h_final[(size_t)b * HID + n] = v;
      }
    }
    __syncthreads();                          // old h fully consumed
    #pragma unroll
    for (int mt = 0; mt < 4; mt++) {
      #pragma unroll
      for (int r = 0; r < 8; r++) {
        int b = mt * 16 + r + hi * 8;
        hsh[(size_t)b * HID + n] = hnew[mt][r];
        if (STORE_SEQ) out_seq[((size_t)b * SEQ + t) * HID + n] = hnew[mt][r];
      }
    }
    __syncthreads();                          // new h visible
  }
}

// ---- tiny classifier head ------------------------------------------------
__global__ void fc_kernel(const float* __restrict__ h1,
                          const float* __restrict__ fcW,
                          const float* __restrict__ fcb,
                          float* __restrict__ out) {
  int tid = blockIdx.x * blockDim.x + threadIdx.x;
  if (tid < BATCH * NCLS) {
    int b = tid >> 3, c = tid & 7;
    const float* hr = h1 + (size_t)b * HID;
    const float* wr = fcW + (size_t)c * HID;
    float s = fcb[c];
    for (int k = 0; k < HID; k++) s += hr[k] * wr[k];
    out[b * NCLS + c] = s;
  }
}

// ---- launcher ------------------------------------------------------------
extern "C" void kernel_launch(void* const* d_in, const int* in_sizes, int n_in,
                              void* d_out, int out_size, void* d_ws, size_t ws_size,
                              hipStream_t stream) {
  (void)in_sizes; (void)n_in; (void)out_size; (void)ws_size;
  const int*   x     = (const int*)  d_in[0];
  const float* emb   = (const float*)d_in[1];
  const float* W_ih0 = (const float*)d_in[2];
  const float* W_hh0 = (const float*)d_in[3];
  const float* b_ih0 = (const float*)d_in[4];
  const float* b_hh0 = (const float*)d_in[5];
  const float* W_ih1 = (const float*)d_in[6];
  const float* W_hh1 = (const float*)d_in[7];
  const float* b_ih1 = (const float*)d_in[8];
  const float* b_hh1 = (const float*)d_in[9];
  const float* fc_W  = (const float*)d_in[10];
  const float* fc_b  = (const float*)d_in[11];
  float* out = (float*)d_out;

  char* ws = (char*)d_ws;
  const size_t W_ELEMS = (size_t)HID * HID;             // 262144
  unsigned short* bfA  = (unsigned short*)(ws);         // 32 MB: xe, reused for out0
  float*          pre  = (float*)(ws + 33554432ull);    // 64 MB: pre0, reused for pre1
  unsigned short* wb   = (unsigned short*)(ws + 100663296ull);  // 2 MB bf16 weights
  float*          h1   = (float*)(ws + 102760448ull);   // 128 KB final hidden
  unsigned short* wih0 = wb;
  unsigned short* whh0 = wb + W_ELEMS;
  unsigned short* wih1 = wb + 2 * W_ELEMS;
  unsigned short* whh1 = wb + 3 * W_ELEMS;

  // embedding gather (f32 -> bf16)
  embed_gather_kernel<<<MROWS, 64, 0, stream>>>(x, emb, bfA);
  // weight conversions
  int cvt_blocks = (int)(W_ELEMS / (256 * 8));          // 128
  cvt_bf16_kernel<<<cvt_blocks, 256, 0, stream>>>(W_ih0, wih0, (int)W_ELEMS);
  cvt_bf16_kernel<<<cvt_blocks, 256, 0, stream>>>(W_hh0, whh0, (int)W_ELEMS);
  cvt_bf16_kernel<<<cvt_blocks, 256, 0, stream>>>(W_ih1, wih1, (int)W_ELEMS);
  cvt_bf16_kernel<<<cvt_blocks, 256, 0, stream>>>(W_hh1, whh1, (int)W_ELEMS);

  // layer 0: input projection + scan (emits out0 bf16 into bfA)
  gemm_inproj_kernel<<<MROWS / 32, 256, 0, stream>>>(bfA, wih0, b_ih0, b_hh0, pre);
  rnn_scan_kernel<true, false><<<1, 1024, 0, stream>>>(pre, whh0, bfA, nullptr);

  // layer 1: input projection + scan (final hidden only)
  gemm_inproj_kernel<<<MROWS / 32, 256, 0, stream>>>(bfA, wih1, b_ih1, b_hh1, pre);
  rnn_scan_kernel<false, true><<<1, 1024, 0, stream>>>(pre, whh1, nullptr, h1);

  // classifier head
  fc_kernel<<<2, 256, 0, stream>>>(h1, fc_W, fc_b, out);
}
